// yourecs_40106404610225
// MI455X (gfx1250) — matrix operand order repaired and checked by reading, not compiled, not run on hardware
//
#include <hip/hip_runtime.h>

// ============================================================================
// GAT 3-layer pipeline for MI455X (gfx1250, wave32, WMMA).
//   GEMMs: v_wmma_f32_16x16x32_f16 (f16 in, f32 acc), one 16x16 C tile/wave.
//   Edge phase: dst-CSR built once per call; segment softmax + aggregation
//   are conflict-free gathers (no float atomics) running out of the 192MB L2.
// ============================================================================

typedef __attribute__((ext_vector_type(16))) _Float16 v16h;
typedef __attribute__((ext_vector_type(8)))  _Float16 v8h;
typedef __attribute__((ext_vector_type(8)))  float    v8f;

#define NEG_SLOPE 0.2f

// ---------------------------------------------------------------------------
// BatchNorm: per-feature mean/var (biased), then normalize -> f16 into the
// concat buffer (columns [0,256) of the [N,1280] layer-2 input).
// ---------------------------------------------------------------------------
__global__ void bn_stats_kernel(const float* __restrict__ x, int N, int F,
                                float* __restrict__ mu, float* __restrict__ rsig) {
  __shared__ float ssum[256];
  __shared__ float ssq[256];
  int c = blockIdx.x;
  float sum = 0.f, sq = 0.f;
  for (int r = threadIdx.x; r < N; r += blockDim.x) {
    float v = x[(size_t)r * F + c];
    sum += v; sq += v * v;
  }
  ssum[threadIdx.x] = sum; ssq[threadIdx.x] = sq;
  __syncthreads();
  for (int s = blockDim.x >> 1; s > 0; s >>= 1) {
    if ((int)threadIdx.x < s) {
      ssum[threadIdx.x] += ssum[threadIdx.x + s];
      ssq[threadIdx.x]  += ssq[threadIdx.x + s];
    }
    __syncthreads();
  }
  if (threadIdx.x == 0) {
    float m   = ssum[0] / (float)N;
    float var = ssq[0] / (float)N - m * m;
    mu[c]   = m;
    rsig[c] = rsqrtf(var + 1e-5f);
  }
}

__global__ void bn_norm_kernel(const float* __restrict__ x,
                               const float* __restrict__ mu,
                               const float* __restrict__ rsig,
                               const float* __restrict__ gamma,
                               const float* __restrict__ beta,
                               _Float16* __restrict__ xcat,
                               int N, int F, int ldcat) {
  size_t i = (size_t)blockIdx.x * blockDim.x + threadIdx.x;
  if (i >= (size_t)N * F) return;
  int r = (int)(i / F), c = (int)(i % F);
  float v = (x[i] - mu[c]) * rsig[c] * gamma[c] + beta[c];
  xcat[(size_t)r * ldcat + c] = (_Float16)v;
}

// ---------------------------------------------------------------------------
// Weight convert+transpose: W [K, Ncols] f32 row-major -> Wt [Ncols, K] f16,
// so each WMMA lane gets one contiguous 32B B-fragment load.
// ---------------------------------------------------------------------------
__global__ void wt_transpose_kernel(const float* __restrict__ W,
                                    _Float16* __restrict__ Wt, int K, int Ncols) {
  size_t i = (size_t)blockIdx.x * blockDim.x + threadIdx.x;
  if (i >= (size_t)K * Ncols) return;
  int k = (int)(i / Ncols), n = (int)(i % Ncols);
  Wt[(size_t)n * K + k] = (_Float16)W[i];
}

__global__ void fill_u32_kernel(unsigned* __restrict__ p, unsigned v, size_t n) {
  size_t i = (size_t)blockIdx.x * blockDim.x + threadIdx.x;
  if (i < n) p[i] = v;
}

// ---------------------------------------------------------------------------
// WMMA GEMM: C[M,Ncols] (f32) = A[M,Kdim] (f16, row-major, stride lda)
//                               x Bt[Ncols,Kdim] (f16, col-major of B).
// 256 threads = 8 waves; each wave owns one 16x16 C tile.
// A-fragment (16x32, contiguous k order): lane holds row M=lane&15;
//   elems 0..15 -> K = k + 16*(lane>>4) + e   (two 16B loads)
// B-fragment (32x16): lane holds col N=lane&15, K = k + 16*(lane>>4) + e
//   (one contiguous 32B load from Bt).
// C/D: elem v -> row rowBase + v + 8*(lane>>4), col colBase + (lane&15).
// ---------------------------------------------------------------------------
__global__ void wmma_gemm_f16(const _Float16* __restrict__ A, int lda,
                              const _Float16* __restrict__ Bt,
                              float* __restrict__ C, int ldc,
                              int Kdim, int Ncols) {
  const int lane = threadIdx.x & 31;
  const int wave = threadIdx.x >> 5;
  const int rowBase = blockIdx.x * 16;
  const int colTile = blockIdx.y * 8 + wave;
  if (colTile * 16 >= Ncols) return;          // wave-uniform: EXEC stays all-1s
  const int colBase = colTile * 16;
  const int hi = lane >> 4;
  const int lo = lane & 15;

  v8f acc = {};
  const _Float16* arow = A  + (size_t)(rowBase + lo) * lda  + (hi << 4);
  const _Float16* brow = Bt + (size_t)(colBase + lo) * Kdim + (hi << 4);

  for (int k = 0; k < Kdim; k += 32) {
    union { v16h v; v8h h[2]; } ua;
    ua.h[0] = *(const v8h*)(arow + k);
    ua.h[1] = *(const v8h*)(arow + k + 8);
    v16h bfrag = *(const v16h*)(brow + k);
    acc = __builtin_amdgcn_wmma_f32_16x16x32_f16(
        /*neg_a=*/false, ua.v, /*neg_b=*/false, bfrag,
        /*c_mod=*/(short)0, acc, /*reuse_a=*/false, /*reuse_b=*/false);
  }

  float* crow = C + (size_t)(rowBase + 8 * hi) * ldc + colBase + lo;
#pragma unroll
  for (int v = 0; v < 8; ++v) crow[(size_t)v * ldc] = acc[v];
}

// ---------------------------------------------------------------------------
// Attention logits: al_src[n,h] = <xw[n,h,:], a_src[h,:]>, same for dst.
// ---------------------------------------------------------------------------
__global__ void att_score_kernel(const float* __restrict__ xw,
                                 const float* __restrict__ a_src,
                                 const float* __restrict__ a_dst,
                                 float* __restrict__ al_s,
                                 float* __restrict__ al_d,
                                 int N, int H, int C) {
  int i = blockIdx.x * blockDim.x + threadIdx.x;
  if (i >= N * H) return;
  int h = i % H;
  const float* row = xw + (size_t)i * C;
  float s = 0.f, d = 0.f;
  for (int c = 0; c < C; ++c) {
    float v = row[c];
    s += v * a_src[h * C + c];
    d += v * a_dst[h * C + c];
  }
  al_s[i] = s; al_d[i] = d;
}

// ---------------------------------------------------------------------------
// CSR build over the virtual edge list of length E+N (idx>=E: self-loops).
// Only cheap integer atomics here; structure is reused by all 3 layers.
// ---------------------------------------------------------------------------
__device__ inline int clampi(int v, int n) { return v < 0 ? 0 : (v >= n ? n - 1 : v); }

__global__ void deg_count_kernel(const int* __restrict__ dst, int E, int N,
                                 int* __restrict__ deg) {
  long long i = (long long)blockIdx.x * blockDim.x + threadIdx.x;
  if (i >= (long long)E + N) return;
  int d = (i < E) ? clampi(dst[i], N) : (int)(i - E);
  atomicAdd(&deg[d], 1);
}

// Single-block chunked Hillis-Steele exclusive scan: rowptr[0..N] from deg.
__global__ void scan_rowptr_kernel(const int* __restrict__ deg,
                                   int* __restrict__ rowptr, int N) {
  __shared__ int tile[256];
  __shared__ int carry;
  const int tid = threadIdx.x;
  if (tid == 0) carry = 0;
  __syncthreads();
  for (int base = 0; base < N; base += 256) {
    int i = base + tid;
    int v = (i < N) ? deg[i] : 0;
    tile[tid] = v;
    __syncthreads();
    for (int off = 1; off < 256; off <<= 1) {
      int t = (tid >= off) ? tile[tid - off] : 0;
      __syncthreads();
      tile[tid] += t;
      __syncthreads();
    }
    int cbase = carry;
    if (i < N) rowptr[i] = cbase + tile[tid] - v;   // exclusive
    __syncthreads();
    if (tid == 255) carry = cbase + tile[255];
    __syncthreads();
  }
  if (tid == 0) rowptr[N] = carry;
}

__global__ void scatter_edges_kernel(const int* __restrict__ src,
                                     const int* __restrict__ dst, int E, int N,
                                     const int* __restrict__ rowptr,
                                     int* __restrict__ cursor,
                                     int* __restrict__ srcid) {
  long long i = (long long)blockIdx.x * blockDim.x + threadIdx.x;
  if (i >= (long long)E + N) return;
  int s = (i < E) ? clampi(src[i], N) : (int)(i - E);
  int d = (i < E) ? clampi(dst[i], N) : (int)(i - E);
  int slot = rowptr[d] + atomicAdd(&cursor[d], 1);
  srcid[slot] = s;
}

// ---------------------------------------------------------------------------
// Segment softmax in CSR order: one thread per (node, head); avg degree ~17.
// alpha[j*H+h] ends up as the normalized attention weight. No atomics.
// ---------------------------------------------------------------------------
__global__ void softmax_csr_kernel(const int* __restrict__ rowptr,
                                   const int* __restrict__ srcid,
                                   const float* __restrict__ al_s,
                                   const float* __restrict__ al_d,
                                   float* __restrict__ alpha, int N, int H) {
  int i = blockIdx.x * blockDim.x + threadIdx.x;
  if (i >= N * H) return;
  int n = i / H, h = i % H;
  int start = rowptr[n], end = rowptr[n + 1];
  float ad = al_d[(size_t)n * H + h];
  float m = -1e30f;
  for (int j = start; j < end; ++j) {
    float ev = al_s[(size_t)srcid[j] * H + h] + ad;
    ev = ev >= 0.f ? ev : NEG_SLOPE * ev;      // leaky_relu
    alpha[(size_t)j * H + h] = ev;
    m = fmaxf(m, ev);
  }
  float ssum = 0.f;
  for (int j = start; j < end; ++j) {
    float ee = __expf(alpha[(size_t)j * H + h] - m);
    alpha[(size_t)j * H + h] = ee;
    ssum += ee;
  }
  float inv = 1.f / (ssum + 1e-16f);
  for (int j = start; j < end; ++j) alpha[(size_t)j * H + h] *= inv;
}

// ---------------------------------------------------------------------------
// Aggregation: one wave per (node, head); lanes stride the channel dim.
// Per edge: coalesced read of xw[src,h,:] (L2-resident), register accumulate,
// one contiguous store. Conflict-free -> no float atomics anywhere.
// ---------------------------------------------------------------------------
__global__ void aggregate_csr_kernel(const int* __restrict__ rowptr,
                                     const int* __restrict__ srcid,
                                     const float* __restrict__ alpha,
                                     const float* __restrict__ xw,
                                     float* __restrict__ out,
                                     int N, int H, int C) {
  const int wid  = blockIdx.x * (blockDim.x >> 5) + (threadIdx.x >> 5);
  const int lane = threadIdx.x & 31;
  if (wid >= N * H) return;                    // wave-uniform
  const int n = wid / H, h = wid % H;
  const int start = rowptr[n], end = rowptr[n + 1];
  float acc0 = 0.f, acc1 = 0.f;                // C <= 64
  for (int j = start; j < end; ++j) {
    float a = alpha[(size_t)j * H + h];
    const float* xr = xw + ((size_t)srcid[j] * H + h) * C;
    if (lane < C)      acc0 += a * xr[lane];
    if (lane + 32 < C) acc1 += a * xr[lane + 32];
  }
  float* orow = out + ((size_t)n * H + h) * C;
  if (lane < C)      orow[lane]      = acc0;
  if (lane + 32 < C) orow[lane + 32] = acc1;
}

// ---------------------------------------------------------------------------
// Finalize (layers 1,2): bias + PReLU(slope from device scalar) -> f16 into
// the next layer's GEMM input at column offset coloff.
// ---------------------------------------------------------------------------
__global__ void finalize_concat_kernel(const float* __restrict__ acc,
                                       const float* __restrict__ bias,
                                       const float* __restrict__ p,
                                       _Float16* __restrict__ dstcat,
                                       int N, int F, int ldcat, int coloff) {
  size_t i = (size_t)blockIdx.x * blockDim.x + threadIdx.x;
  if (i >= (size_t)N * F) return;
  int r = (int)(i / F), c = (int)(i % F);
  float v = acc[i] + bias[c];
  float slope = *p;
  v = v >= 0.f ? v : slope * v;
  dstcat[(size_t)r * ldcat + coloff + c] = (_Float16)v;
}

// ---------------------------------------------------------------------------
// Finalize layer 3: mean over 8 heads + b3, log_softmax over 16 classes.
// ---------------------------------------------------------------------------
__global__ void finalize_out_kernel(const float* __restrict__ acc,
                                    const float* __restrict__ b3,
                                    float* __restrict__ out, int N) {
  int n = blockIdx.x * blockDim.x + threadIdx.x;
  if (n >= N) return;
  float logits[16];
  for (int c = 0; c < 16; ++c) {
    float s = 0.f;
    for (int h = 0; h < 8; ++h) s += acc[((size_t)n * 8 + h) * 16 + c];
    logits[c] = s * 0.125f + b3[c];
  }
  float mx = logits[0];
  for (int c = 1; c < 16; ++c) mx = fmaxf(mx, logits[c]);
  float se = 0.f;
  for (int c = 0; c < 16; ++c) se += __expf(logits[c] - mx);
  float lse = mx + __logf(se);
  for (int c = 0; c < 16; ++c) out[(size_t)n * 16 + c] = logits[c] - lse;
}

// ===========================================================================
// Host side
// ===========================================================================
static inline unsigned cdiv(size_t a, size_t b) { return (unsigned)((a + b - 1) / b); }

extern "C" void kernel_launch(void* const* d_in, const int* in_sizes, int n_in,
                              void* d_out, int out_size, void* d_ws, size_t ws_size,
                              hipStream_t stream) {
  const float* x     = (const float*)d_in[0];
  const int*   ei    = (const int*)d_in[1];    // [2,E]; harness delivers int32
  const float* gamma = (const float*)d_in[2];
  const float* beta  = (const float*)d_in[3];
  const float* W1    = (const float*)d_in[4];
  const float* a1s   = (const float*)d_in[5];
  const float* a1d   = (const float*)d_in[6];
  const float* b1    = (const float*)d_in[7];
  const float* p1    = (const float*)d_in[8];
  const float* W2    = (const float*)d_in[9];
  const float* a2s   = (const float*)d_in[10];
  const float* a2d   = (const float*)d_in[11];
  const float* b2    = (const float*)d_in[12];
  const float* p2    = (const float*)d_in[13];
  const float* W3    = (const float*)d_in[14];
  const float* a3s   = (const float*)d_in[15];
  const float* a3d   = (const float*)d_in[16];
  const float* b3    = (const float*)d_in[17];

  const int Fin = 256;
  const int H1 = 16, C1 = 64,  F1 = H1 * C1;     // 1024
  const int F2in = Fin + F1;                      // 1280
  const int H2 = 8,  C2 = 8,   F2 = H2 * C2;     // 64
  const int H3 = 8,  C3 = 16,  F3 = H3 * C3;     // 128
  const int N = in_sizes[0] / Fin;                // 20000 (multiple of 16)
  const int E = in_sizes[1] / 2;                  // 320000
  const int Etot = E + N;                         // + self loops
  const int* src = ei;
  const int* dst = ei + E;

  // ---- workspace layout (256B-aligned bump allocator) ----
  char* w = (char*)d_ws;
  auto alloc = [&](size_t bytes) -> void* {
    void* p = (void*)w;
    w += (bytes + 255) & ~(size_t)255;
    return p;
  };
  _Float16* xcat = (_Float16*)alloc((size_t)N * F2in * 2);  // BN(x) ++ h1, f16
  _Float16* x3   = (_Float16*)alloc((size_t)N * F2 * 2);    // h2, f16
  _Float16* W1t  = (_Float16*)alloc((size_t)F1 * Fin * 2);
  _Float16* W2t  = (_Float16*)alloc((size_t)F2 * F2in * 2);
  _Float16* W3t  = (_Float16*)alloc((size_t)F3 * F2 * 2);
  float* mu     = (float*)alloc((size_t)Fin * 4);
  float* rsig   = (float*)alloc((size_t)Fin * 4);
  float* xw     = (float*)alloc((size_t)N * F1 * 4);        // reused, max [N,1024]
  float* agg    = (float*)alloc((size_t)N * F1 * 4);        // reused, max [N,1024]
  float* al_s   = (float*)alloc((size_t)N * H1 * 4);
  float* al_d   = (float*)alloc((size_t)N * H1 * 4);
  int*   rowptr = (int*)alloc((size_t)(N + 1) * 4);
  int*   cursor = (int*)alloc((size_t)N * 4);               // deg, then cursor
  int*   srcid  = (int*)alloc((size_t)Etot * 4);
  float* alpha  = (float*)alloc((size_t)Etot * H1 * 4);

  const int T = 256;

  // ---- BatchNorm -> f16 columns [0,256) of xcat ----
  bn_stats_kernel<<<Fin, T, 0, stream>>>(x, N, Fin, mu, rsig);
  bn_norm_kernel<<<cdiv((size_t)N * Fin, T), T, 0, stream>>>(x, mu, rsig, gamma, beta,
                                                             xcat, N, Fin, F2in);
  // ---- weights -> f16 col-major ----
  wt_transpose_kernel<<<cdiv((size_t)Fin * F1, T), T, 0, stream>>>(W1, W1t, Fin, F1);
  wt_transpose_kernel<<<cdiv((size_t)F2in * F2, T), T, 0, stream>>>(W2, W2t, F2in, F2);
  wt_transpose_kernel<<<cdiv((size_t)F2 * F3, T), T, 0, stream>>>(W3, W3t, F2, F3);

  // ---- dst-CSR build (shared by all layers) ----
  fill_u32_kernel<<<cdiv((size_t)N, T), T, 0, stream>>>((unsigned*)cursor, 0u, (size_t)N);
  deg_count_kernel<<<cdiv((size_t)Etot, T), T, 0, stream>>>(dst, E, N, cursor);
  scan_rowptr_kernel<<<1, T, 0, stream>>>(cursor, rowptr, N);
  fill_u32_kernel<<<cdiv((size_t)N, T), T, 0, stream>>>((unsigned*)cursor, 0u, (size_t)N);
  scatter_edges_kernel<<<cdiv((size_t)Etot, T), T, 0, stream>>>(src, dst, E, N,
                                                                rowptr, cursor, srcid);

  auto run_layer = [&](const _Float16* Ain, int lda, const _Float16* Bt, int Kdim,
                       int H, int C, const float* asrc, const float* adst) {
    const int Fo = H * C;
    wmma_gemm_f16<<<dim3((N + 15) / 16, cdiv(Fo, 128)), T, 0, stream>>>(
        Ain, lda, Bt, xw, Fo, Kdim, Fo);
    att_score_kernel<<<cdiv((size_t)N * H, T), T, 0, stream>>>(
        xw, asrc, adst, al_s, al_d, N, H, C);
    softmax_csr_kernel<<<cdiv((size_t)N * H, T), T, 0, stream>>>(
        rowptr, srcid, al_s, al_d, alpha, N, H);
    aggregate_csr_kernel<<<cdiv((size_t)N * H, 8), T, 0, stream>>>(
        rowptr, srcid, alpha, xw, agg, N, H, C);
  };

  // ---- Layer 1: [N,256] -> [N,1024], concat+PReLU into xcat[:,256:1280] ----
  run_layer(xcat, F2in, W1t, Fin, H1, C1, a1s, a1d);
  finalize_concat_kernel<<<cdiv((size_t)N * F1, T), T, 0, stream>>>(
      agg, b1, p1, xcat, N, F1, F2in, Fin);

  // ---- Layer 2: [N,1280] -> [N,64], PReLU -> x3 ----
  run_layer(xcat, F2in, W2t, F2in, H2, C2, a2s, a2d);
  finalize_concat_kernel<<<cdiv((size_t)N * F2, T), T, 0, stream>>>(
      agg, b2, p2, x3, N, F2, F2, 0);

  // ---- Layer 3: [N,64] -> [N,8,16], mean heads + b3 + log_softmax ----
  run_layer(x3, F2, W3t, F2, H3, C3, a3s, a3d);
  finalize_out_kernel<<<cdiv((size_t)N, T), T, 0, stream>>>(agg, b3, (float*)d_out, N);
}
